// GraphAgentNetwork_46385646797012
// MI455X (gfx1250) — matrix operand order repaired and checked by reading, not compiled, not run on hardware
//
#include <hip/hip_runtime.h>
#include <hip/hip_bf16.h>
#include <stdint.h>

#define OBS   128
#define HID   64
#define HEADS 4
#define ACT   32

typedef __attribute__((ext_vector_type(16))) __bf16 v16bf;
typedef __attribute__((ext_vector_type(8)))  __bf16 v8bf;
typedef __attribute__((ext_vector_type(8)))  float  v8f;

// ---------------------------------------------------------------- utilities
__global__ void k_f32_to_bf16(const float* __restrict__ in, __bf16* __restrict__ out,
                              long long n) {
  long long i = (long long)blockIdx.x * blockDim.x + threadIdx.x;
  if (i < n) out[i] = (__bf16)in[i];
}

__device__ inline void atomicMaxF(float* addr, float val) {
  // sign-split trick: correct combined float max with mixed-sign updates
  if (val >= 0.0f) atomicMax((int*)addr, __float_as_int(val));
  else             atomicMin((unsigned int*)addr, __float_as_uint(val));
}

// ---------------------------------------------- B-weight fragment packing
// Fuses f32->bf16 with reordering B[K,Nc] into WMMA fragment order:
//   Bp[((tn*(K/32) + ks)*32 + lane)*16 + j] = B[ks*32 + (lane>>4)*8 + (j&7) + (j>>3)*16][tn*16 + (lane&15)]
// so each lane's 16-element B fragment is one contiguous, 32B-aligned chunk.
__global__ void k_pack_b(const float* __restrict__ B, __bf16* __restrict__ Bp,
                         int K, int Nc) {
  int idx = blockIdx.x * blockDim.x + threadIdx.x;
  if (idx >= K * Nc) return;
  int j    = idx & 15;
  int lane = (idx >> 4) & 31;
  int ks   = (idx >> 9) % (K >> 5);
  int tn   = (idx >> 9) / (K >> 5);
  int lm = lane & 15, hi = lane >> 4;
  int k = ks * 32 + hi * 8 + (j & 7) + (j >> 3) * 16;
  int n = tn * 16 + lm;
  Bp[idx] = (__bf16)B[(size_t)k * Nc + n];
}

// --------------------------------------------------- WMMA GEMM (bf16 -> f32)
// C[M,Nc] = A[M,K] * Bp (+bias, optional relu). One wave per 16x16 tile.
// A-frag: two aligned 16B vector loads; B-frag: one contiguous 32B load.
__global__ void k_gemm_wmma(const __bf16* __restrict__ A, const __bf16* __restrict__ Bp,
                            const float* __restrict__ bias, float* __restrict__ C,
                            int M, int K, int Nc, int relu) {
  int wave = (int)((blockIdx.x * blockDim.x + threadIdx.x) >> 5);
  int lane = threadIdx.x & 31;
  int tiles_n = Nc >> 4;
  int tiles   = (M >> 4) * tiles_n;
  if (wave >= tiles) return;             // wave-uniform: EXEC all-1s for WMMA
  int tm = wave / tiles_n, tn = wave % tiles_n;
  int lm = lane & 15, hi = lane >> 4;
  int kSteps = K >> 5;

  v8f acc = {};
  // A row for this lane, as 8-element bf16 vectors (16B aligned: K%32==0)
  const v8bf* arow = (const v8bf*)(A + (size_t)(tm * 16 + lm) * K) + hi;
  const v16bf* bfrag = (const v16bf*)Bp + (size_t)(tn * kSteps) * 32 + lane;
  for (int ks = 0; ks < kSteps; ++ks) {
    v8bf a0 = arow[ks * 4];          // K = {hi*8 .. hi*8+7}
    v8bf a1 = arow[ks * 4 + 2];      // K = {16+hi*8 .. 16+hi*8+7}
    v16bf af = __builtin_shufflevector(a0, a1, 0, 1, 2, 3, 4, 5, 6, 7,
                                       8, 9, 10, 11, 12, 13, 14, 15);
    v16bf bm = bfrag[ks * 32];       // lane's 16 B-elements, contiguous 32B
    acc = __builtin_amdgcn_wmma_f32_16x16x32_bf16(false, af, false, bm,
                                                  (short)0, acc, false, false);
  }
  int n = tn * 16 + lm;
  float bv = bias ? bias[n] : 0.0f;
#pragma unroll
  for (int r = 0; r < 8; ++r) {          // C: vgpr r = row (hi*8 + r), col = lane
    int m = tm * 16 + hi * 8 + r;
    float v = acc[r] + bv;
    if (relu) v = fmaxf(v, 0.0f);
    C[(size_t)m * Nc + n] = v;
  }
}

// -------------------------------------------------------- attention scores
// One wave per node: 256-float row read coalesced, 8-lane group per head.
__global__ void k_attn_scores(const float* __restrict__ xl,
                              const float* __restrict__ att_src,
                              const float* __restrict__ att_dst,
                              float* __restrict__ a_src, float* __restrict__ a_dst,
                              int Nn) {
  int wave = (int)((blockIdx.x * blockDim.x + threadIdx.x) >> 5);
  int lane = threadIdx.x & 31;
  if (wave >= Nn) return;
  const float* row = xl + (size_t)wave * (HEADS * HID);
  int h = lane >> 3;
  int base = lane * 8;
  int ib = base & (HID - 1);
  float ss = 0.0f, sd = 0.0f;
#pragma unroll
  for (int j = 0; j < 8; ++j) {
    float v = row[base + j];
    ss += v * att_src[h * HID + ib + j];
    sd += v * att_dst[h * HID + ib + j];
  }
  for (int off = 4; off > 0; off >>= 1) {
    ss += __shfl_down(ss, off, 8);
    sd += __shfl_down(sd, off, 8);
  }
  if ((lane & 7) == 0) {
    a_src[wave * HEADS + h] = ss;
    a_dst[wave * HEADS + h] = sd;
  }
}

__global__ void k_gat_init(float* outac, float* mbuf, float* sbuf, int Nn) {
  long long i = (long long)blockIdx.x * blockDim.x + threadIdx.x;
  if (i < (long long)Nn * HEADS * HID) outac[i] = 0.0f;
  if (i < (long long)Nn * HEADS) { mbuf[i] = -3.0e38f; sbuf[i] = 0.0f; }
}

// pass1: alpha = leaky_relu(a_src[src]+a_dst[dst]); segment max over dst
__global__ void k_edge_pass1(const long long* __restrict__ ei, int E_, int Nn,
                             const float* __restrict__ a_src,
                             const float* __restrict__ a_dst,
                             float* __restrict__ alpha, float* __restrict__ mbuf) {
  long long e = (long long)blockIdx.x * blockDim.x + threadIdx.x;
  long long Et = (long long)E_ + Nn;
  if (e >= Et) return;
  int s, d;
  if (e < E_) { s = (int)ei[e]; d = (int)ei[(long long)E_ + e]; }
  else        { s = d = (int)(e - E_); }
#pragma unroll
  for (int h = 0; h < HEADS; ++h) {
    float a = a_src[s * HEADS + h] + a_dst[d * HEADS + h];
    a = (a > 0.0f) ? a : 0.2f * a;
    alpha[e * HEADS + h] = a;
    atomicMaxF(mbuf + (size_t)d * HEADS + h, a);
  }
}

// pass2: p = exp(alpha - m[dst]); segment sum over dst (p overwrites alpha)
__global__ void k_edge_pass2(const long long* __restrict__ ei, int E_, int Nn,
                             float* __restrict__ alpha,
                             const float* __restrict__ mbuf, float* __restrict__ sbuf) {
  long long e = (long long)blockIdx.x * blockDim.x + threadIdx.x;
  long long Et = (long long)E_ + Nn;
  if (e >= Et) return;
  int d;
  if (e < E_) d = (int)ei[(long long)E_ + e];
  else        d = (int)(e - E_);
#pragma unroll
  for (int h = 0; h < HEADS; ++h) {
    float p = __expf(alpha[e * HEADS + h] - mbuf[(size_t)d * HEADS + h]);
    alpha[e * HEADS + h] = p;
    atomicAdd(sbuf + (size_t)d * HEADS + h, p);
  }
}

// pass3: out[dst,h,:] += xl[src,h,:] * (p / s[dst,h]); one wave per edge
__global__ void k_edge_pass3(const long long* __restrict__ ei, int E_, int Nn,
                             const float* __restrict__ xl, const float* __restrict__ p,
                             const float* __restrict__ sbuf, float* __restrict__ outac) {
  long long wv = (long long)blockIdx.x * (blockDim.x >> 5) + (threadIdx.x >> 5);
  int lane = threadIdx.x & 31;
  long long Et = (long long)E_ + Nn;
  if (wv >= Et) return;
  int s, d;
  if (wv < E_) { s = (int)ei[wv]; d = (int)ei[(long long)E_ + wv]; }
  else         { s = d = (int)(wv - E_); }
  int h = lane >> 3;
  float w = p[wv * HEADS + h] / sbuf[(size_t)d * HEADS + h];
  const float* srow = xl + (size_t)s * (HEADS * HID) + lane * 8;
  float*       drow = outac + (size_t)d * (HEADS * HID) + lane * 8;
#pragma unroll
  for (int j = 0; j < 8; ++j) atomicAdd(drow + j, srow[j] * w);
}

// head-mean + bias + relu
__global__ void k_gat_finalize(const float* __restrict__ outac,
                               const float* __restrict__ bias,
                               float* __restrict__ x, int Nn) {
  long long idx = (long long)blockIdx.x * blockDim.x + threadIdx.x;
  if (idx >= (long long)Nn * HID) return;
  int n = (int)(idx >> 6), i = (int)(idx & (HID - 1));
  const float* r = outac + (size_t)n * (HEADS * HID);
  float v = 0.25f * (r[i] + r[HID + i] + r[2 * HID + i] + r[3 * HID + i]) + bias[i];
  x[idx] = fmaxf(v, 0.0f);
}

// -------------------------------------------------------------------- host
extern "C" void kernel_launch(void* const* d_in, const int* in_sizes, int n_in,
                              void* d_out, int out_size, void* d_ws, size_t ws_size,
                              hipStream_t stream) {
  const float*     obs    = (const float*)d_in[0];
  const long long* ei     = (const long long*)d_in[1];   // int64 per reference
  const float*     enc_w1 = (const float*)d_in[2];
  const float*     enc_b1 = (const float*)d_in[3];
  const float*     enc_w2 = (const float*)d_in[4];
  const float*     enc_b2 = (const float*)d_in[5];
  const float*     dec_w1 = (const float*)d_in[14];
  const float*     dec_b1 = (const float*)d_in[15];
  const float*     dec_w2 = (const float*)d_in[16];
  const float*     dec_b2 = (const float*)d_in[17];

  const int Nn = in_sizes[0] / OBS;      // 100000
  const int E_ = in_sizes[1] / 2;        // 800000
  const long long Et = (long long)E_ + Nn;

  // workspace carve (~277 MB total)
  float*  xl    = (float*)d_ws;                           // [N,256]
  float*  outac = xl    + (size_t)Nn * HEADS * HID;       // [N,256]
  float*  xbuf  = outac + (size_t)Nn * HEADS * HID;       // [N,64]
  float*  asrc  = xbuf  + (size_t)Nn * HID;               // [N,4]
  float*  adst  = asrc  + (size_t)Nn * HEADS;             // [N,4]
  float*  mbuf  = adst  + (size_t)Nn * HEADS;             // [N,4]
  float*  sbuf  = mbuf  + (size_t)Nn * HEADS;             // [N,4]
  float*  alph  = sbuf  + (size_t)Nn * HEADS;             // [E',4]
  __bf16* bfX   = (__bf16*)(alph + (size_t)Et * HEADS);   // [N,128] bf16
  __bf16* bfW   = bfX + (size_t)Nn * OBS;                 // up to 64*256 bf16 (packed)
  float*  tmp0  = xl;  // alias: xl only live inside a GAT layer

  auto cvt = [&](const float* src, __bf16* dst, long long n) {
    k_f32_to_bf16<<<(unsigned)((n + 255) / 256), 256, 0, stream>>>(src, dst, n);
  };
  auto packB = [&](const float* B, int K, int Nc) {
    k_pack_b<<<(K * Nc + 255) / 256, 256, 0, stream>>>(B, bfW, K, Nc);
  };
  auto gemm = [&](const __bf16* A, const float* bias, float* C,
                  int M, int K, int Nc, int relu) {
    int tiles = (M / 16) * (Nc / 16);
    k_gemm_wmma<<<(tiles + 7) / 8, 256, 0, stream>>>(A, bfW, bias, C, M, K, Nc, relu);
  };

  // ---- encoder: x = relu(obs@W1+b1)@W2+b2
  cvt(obs, bfX, (long long)Nn * OBS);
  packB(enc_w1, OBS, HID);
  gemm(bfX, enc_b1, tmp0, Nn, OBS, HID, 1);
  cvt(tmp0, bfX, (long long)Nn * HID);
  packB(enc_w2, HID, HID);
  gemm(bfX, enc_b2, xbuf, Nn, HID, HID, 0);

  // ---- two GAT layers
  for (int l = 0; l < 2; ++l) {
    const float* gw  = (const float*)d_in[6 + l * 4];
    const float* gas = (const float*)d_in[7 + l * 4];
    const float* gad = (const float*)d_in[8 + l * 4];
    const float* gb  = (const float*)d_in[9 + l * 4];

    cvt(xbuf, bfX, (long long)Nn * HID);
    packB(gw, HID, HEADS * HID);
    gemm(bfX, nullptr, xl, Nn, HID, HEADS * HID, 0);        // xl = x @ W

    k_attn_scores<<<(Nn * 32 + 255) / 256, 256, 0, stream>>>(xl, gas, gad, asrc, adst, Nn);
    long long initN = (long long)Nn * HEADS * HID;
    k_gat_init<<<(unsigned)((initN + 255) / 256), 256, 0, stream>>>(outac, mbuf, sbuf, Nn);
    k_edge_pass1<<<(unsigned)((Et + 255) / 256), 256, 0, stream>>>(ei, E_, Nn, asrc, adst, alph, mbuf);
    k_edge_pass2<<<(unsigned)((Et + 255) / 256), 256, 0, stream>>>(ei, E_, Nn, alph, mbuf, sbuf);
    k_edge_pass3<<<(unsigned)((Et + 7) / 8), 256, 0, stream>>>(ei, E_, Nn, xl, alph, sbuf, outac);
    k_gat_finalize<<<(unsigned)(((long long)Nn * HID + 255) / 256), 256, 0, stream>>>(outac, gb, xbuf, Nn);
  }

  // ---- decoder: q = relu(x@W1+b1)@W2+b2
  cvt(xbuf, bfX, (long long)Nn * HID);
  packB(dec_w1, HID, HID);
  gemm(bfX, dec_b1, tmp0, Nn, HID, HID, 1);
  cvt(tmp0, bfX, (long long)Nn * HID);
  packB(dec_w2, HID, ACT);
  gemm(bfX, dec_b2, (float*)d_out, Nn, HID, ACT, 0);
}
